// DenoiseNet_28767690949312
// MI455X (gfx1250) — matrix operand order repaired and checked by reading, not compiled, not run on hardware
//
#include <hip/hip_runtime.h>
#include <hip/hip_bf16.h>

// ---------------- problem constants ----------------
#define BB        4
#define N_NOISY   4096
#define N_CLEANPT 4500
#define N_TRAIN   128
#define FRAME_K   32
#define CLEAN_C   4
#define FEAT_K    16
#define FE_HID    64
#define FEAT_DIM  128
#define HID       128
#define NBLK      4
#define CDIM      131          // FEAT_DIM + 3
#define KPAD      160          // CDIM padded to multiple of 32 for WMMA K
#define NPTS      (BB * N_TRAIN)          // 512 sampled points
#define MROWS     (NPTS * FRAME_K)        // 16384 score-net rows

typedef __attribute__((ext_vector_type(8)))  _Float16 v8h;
typedef __attribute__((ext_vector_type(16))) _Float16 v16h;
typedef __attribute__((ext_vector_type(8)))  float    v8f;

// ---------------- fragment helpers ----------------
__device__ __forceinline__ v16h pack16(v8h lo, v8h hi) {
  v16h r;
#pragma unroll
  for (int j = 0; j < 8; ++j) { r[j] = lo[j]; r[j + 8] = hi[j]; }
  return r;
}
// A-matrix fragment: per ISA, lane half holds K = base+[0..7] and base+16+[0..7]
__device__ __forceinline__ v16h frag_a(const _Float16* p) {
  return pack16(*(const v8h*)p, *(const v8h*)(p + 16));
}
// B-matrix fragment: lane half holds K = base+[0..7] and base+16+[0..7] for one output column
__device__ __forceinline__ v16h frag_b(const _Float16* p) {
  return pack16(*(const v8h*)p, *(const v8h*)(p + 16));
}
__device__ __forceinline__ v8f vzero8() {
  v8f z;
#pragma unroll
  for (int j = 0; j < 8; ++j) z[j] = 0.0f;
  return z;
}

// ---------------- kernel 1: 16-NN (3D) for the 512 sampled points ----------------
__global__ __launch_bounds__(128) void knn16_kernel(const float* __restrict__ pcl,
                                                    const int* __restrict__ pidx,
                                                    int* __restrict__ out) {
  __shared__ float sx[128], sy[128], sz[128];
  const int tid = threadIdx.x;
  const int b = blockIdx.x;            // one batch per block
  const int t = b * N_TRAIN + tid;     // global sampled-point id
  const float* P = pcl + (size_t)b * N_NOISY * 3;
  const int q = pidx[tid];
  const float qx = P[q * 3 + 0], qy = P[q * 3 + 1], qz = P[q * 3 + 2];

  float bd[FEAT_K]; int bi[FEAT_K];
#pragma unroll
  for (int s = 0; s < FEAT_K; ++s) { bd[s] = 3.4e38f; bi[s] = 0; }

  for (int tile = 0; tile < N_NOISY; tile += 128) {
    __syncthreads();
    const int r = tile + tid;
    sx[tid] = P[r * 3 + 0]; sy[tid] = P[r * 3 + 1]; sz[tid] = P[r * 3 + 2];
    __syncthreads();
    for (int u = 0; u < 128; ++u) {
      const float dx = sx[u] - qx, dy = sy[u] - qy, dz = sz[u] - qz;
      const float d = dx * dx + dy * dy + dz * dz;
      if (d < bd[FEAT_K - 1]) {
        bd[FEAT_K - 1] = d; bi[FEAT_K - 1] = tile + u;
#pragma unroll
        for (int s = FEAT_K - 1; s > 0; --s) {
          if (bd[s] < bd[s - 1]) {
            float td = bd[s]; bd[s] = bd[s - 1]; bd[s - 1] = td;
            int   ti = bi[s]; bi[s] = bi[s - 1]; bi[s - 1] = ti;
          }
        }
      }
    }
  }
#pragma unroll
  for (int s = 0; s < FEAT_K; ++s) out[t * FEAT_K + s] = bi[s];
}

// ---------------- kernel 2: edge MLP feature net (only 512 points) ----------------
__global__ __launch_bounds__(128) void featnet_kernel(const float* __restrict__ pcl,
                                                      const int* __restrict__ pidx,
                                                      const int* __restrict__ knn,
                                                      const float* __restrict__ feW1,
                                                      const float* __restrict__ feb1,
                                                      const float* __restrict__ feW2,
                                                      const float* __restrict__ feb2,
                                                      float* __restrict__ feat) {
  __shared__ float nb[FEAT_K][3];
  __shared__ float ctr[3];
  __shared__ float h[FE_HID];
  const int pt = blockIdx.x;                 // 0..511
  const int b = pt >> 7, j = pt & 127;
  const float* P = pcl + (size_t)b * N_NOISY * 3;
  const int tid = threadIdx.x;

  if (tid < FEAT_K) {
    const int r = knn[pt * FEAT_K + tid];
    nb[tid][0] = P[r * 3 + 0]; nb[tid][1] = P[r * 3 + 1]; nb[tid][2] = P[r * 3 + 2];
  }
  if (tid == 127) {
    const int q = pidx[j];
    ctr[0] = P[q * 3 + 0]; ctr[1] = P[q * 3 + 1]; ctr[2] = P[q * 3 + 2];
  }
  __syncthreads();

  if (tid < FE_HID) {
    const float cx = ctr[0], cy = ctr[1], cz = ctr[2];
    const float w0 = feW1[0 * FE_HID + tid], w1 = feW1[1 * FE_HID + tid],
                w2 = feW1[2 * FE_HID + tid], w3 = feW1[3 * FE_HID + tid],
                w4 = feW1[4 * FE_HID + tid], w5 = feW1[5 * FE_HID + tid];
    const float bias = feb1[tid];
    float m = 0.0f;                          // relu outputs are >= 0
    for (int n = 0; n < FEAT_K; ++n) {
      const float ex = nb[n][0] - cx, ey = nb[n][1] - cy, ez = nb[n][2] - cz;
      float v = bias + cx * w0 + cy * w1 + cz * w2 + ex * w3 + ey * w4 + ez * w5;
      v = v > 0.0f ? v : 0.0f;
      m = v > m ? v : m;
    }
    h[tid] = m;
  }
  __syncthreads();

  float s = feb2[tid];
  for (int k2 = 0; k2 < FE_HID; ++k2) s += h[k2] * feW2[k2 * FEAT_DIM + tid];
  feat[(size_t)pt * FEAT_DIM + tid] = s > 0.0f ? s : 0.0f;
}

// ---------------- kernel 3: 32-NN (xy) frames ----------------
__global__ __launch_bounds__(128) void frameknn_kernel(const float* __restrict__ pcl,
                                                       const int* __restrict__ pidx,
                                                       float* __restrict__ frames) {
  __shared__ float sx[128], sy[128];
  const int tid = threadIdx.x;
  const int b = blockIdx.x;
  const int t = b * N_TRAIN + tid;
  const float* P = pcl + (size_t)b * N_NOISY * 3;
  const int q = pidx[tid];
  const float qx = P[q * 3 + 0], qy = P[q * 3 + 1];

  float bd[FRAME_K]; int bi[FRAME_K];
#pragma unroll
  for (int s = 0; s < FRAME_K; ++s) { bd[s] = 3.4e38f; bi[s] = 0; }

  for (int tile = 0; tile < N_NOISY; tile += 128) {
    __syncthreads();
    const int r = tile + tid;
    sx[tid] = P[r * 3 + 0]; sy[tid] = P[r * 3 + 1];
    __syncthreads();
    for (int u = 0; u < 128; ++u) {
      const float dx = sx[u] - qx, dy = sy[u] - qy;
      const float d = dx * dx + dy * dy;
      if (d < bd[FRAME_K - 1]) {
        bd[FRAME_K - 1] = d; bi[FRAME_K - 1] = tile + u;
#pragma unroll
        for (int s = FRAME_K - 1; s > 0; --s) {
          if (bd[s] < bd[s - 1]) {
            float td = bd[s]; bd[s] = bd[s - 1]; bd[s - 1] = td;
            int   ti = bi[s]; bi[s] = bi[s - 1]; bi[s - 1] = ti;
          }
        }
      }
    }
  }
#pragma unroll
  for (int s = 0; s < FRAME_K; ++s) {
    const int r = bi[s];
    float* o = frames + ((size_t)t * FRAME_K + s) * 3;
    o[0] = P[r * 3 + 0]; o[1] = P[r * 3 + 1]; o[2] = P[r * 3 + 2];
  }
}

// ---------------- kernel 4: 4-NN (xy) vs clean cloud -> grad_target ----------------
__global__ __launch_bounds__(256) void cleanknn_kernel(const float* __restrict__ pcl_clean,
                                                       const float* __restrict__ frames,
                                                       float* __restrict__ target) {
  __shared__ float sx[256], sy[256], sz[256];
  const int tid = threadIdx.x;
  const int i = blockIdx.x * 256 + tid;      // 0..16383 (16 blocks per batch)
  const int b = blockIdx.x >> 4;
  const float* C = pcl_clean + (size_t)b * N_CLEANPT * 3;
  const float fx = frames[i * 3 + 0], fy = frames[i * 3 + 1], fz = frames[i * 3 + 2];

  float bd[CLEAN_C], bz[CLEAN_C];
#pragma unroll
  for (int s = 0; s < CLEAN_C; ++s) { bd[s] = 3.4e38f; bz[s] = 0.0f; }

  for (int tile = 0; tile < N_CLEANPT; tile += 256) {
    __syncthreads();
    const int r = tile + tid;
    if (r < N_CLEANPT) { sx[tid] = C[r * 3 + 0]; sy[tid] = C[r * 3 + 1]; sz[tid] = C[r * 3 + 2]; }
    __syncthreads();
    const int lim = (N_CLEANPT - tile) < 256 ? (N_CLEANPT - tile) : 256;
    for (int u = 0; u < lim; ++u) {
      const float dx = sx[u] - fx, dy = sy[u] - fy;
      const float d = dx * dx + dy * dy;
      if (d < bd[CLEAN_C - 1]) {
        bd[CLEAN_C - 1] = d; bz[CLEAN_C - 1] = sz[u];
#pragma unroll
        for (int s = CLEAN_C - 1; s > 0; --s) {
          if (bd[s] < bd[s - 1]) {
            float td = bd[s]; bd[s] = bd[s - 1]; bd[s - 1] = td;
            float tz = bz[s]; bz[s] = bz[s - 1]; bz[s - 1] = tz;
          }
        }
      }
    }
  }
  // grad_target = -noise_z = mean(clean_z) - frame_z
  target[i] = 0.25f * (bz[0] + bz[1] + bz[2] + bz[3]) - fz;
}

// ---------------- kernel 5: f32 weights -> f16, transposed + K-padded ----------------
__global__ void wt_convert(const float* __restrict__ W, _Float16* __restrict__ WT,
                           int K, int N, int Kpad) {
  const int i = blockIdx.x * blockDim.x + threadIdx.x;
  if (i >= N * Kpad) return;
  const int n = i / Kpad, k = i - n * Kpad;
  const float v = (k < K) ? W[k * N + n] : 0.0f;
  WT[i] = (_Float16)v;
}

// ---------------- kernel 6: pack X = [frames_centered | feat | pad] as f16 ----------------
__global__ __launch_bounds__(192) void pack_x_kernel(const float* __restrict__ pcl,
                                                     const int* __restrict__ pidx,
                                                     const float* __restrict__ frames,
                                                     const float* __restrict__ feat,
                                                     _Float16* __restrict__ Xh) {
  const int row = blockIdx.x;            // 0..16383
  const int c = threadIdx.x;
  if (c >= KPAD) return;
  const int pt = row >> 5, kk = row & 31;
  const int b = pt >> 7, j = pt & 127;
  float v = 0.0f;
  if (c < 3) {
    const int q = pidx[j];
    const float cv = pcl[((size_t)b * N_NOISY + q) * 3 + c];
    const float fv = frames[((size_t)pt * FRAME_K + kk) * 3 + c];
    v = fv - cv;
  } else if (c < 3 + FEAT_DIM) {
    v = feat[(size_t)pt * FEAT_DIM + (c - 3)];
  }
  Xh[(size_t)row * KPAD + c] = (_Float16)v;
}

// ---------------- kernel 7: fused score net, WMMA f16 -> f32 ----------------
__global__ __launch_bounds__(256) void score_wmma_kernel(
    const _Float16* __restrict__ Xh,
    const _Float16* __restrict__ WpT,   // [HID][KPAD]
    const _Float16* __restrict__ W0T,   // [NBLK][HID][HID]
    const _Float16* __restrict__ W1T,   // [NBLK][HID][HID]
    const _Float16* __restrict__ WcT,   // [NBLK][HID][KPAD]
    const float* __restrict__ bp,  const float* __restrict__ g0, const float* __restrict__ b0,
    const float* __restrict__ bf0, const float* __restrict__ g1, const float* __restrict__ b1,
    const float* __restrict__ bf1, const float* __restrict__ bc,
    const float* __restrict__ gO,  const float* __restrict__ bO,
    const float* __restrict__ Wo,  const float* __restrict__ bo,
    float* __restrict__ pred) {
  __shared__ _Float16 sAct[64][HID];   // 16 KB: current layer input (f16)
  __shared__ float    sNet[64][HID];   // 32 KB: residual stream (f32)

  const int tid  = threadIdx.x;
  const int wave = tid >> 5, lane = tid & 31;
  const int mt   = wave >> 1;           // row tile 0..3
  const int ntB  = (wave & 1) * 4;      // col tile base (4 tiles per wave)
  const int laneR = lane & 15, laneH = lane >> 4;
  const int rowBase = blockIdx.x * 64;
  const int aRow = rowBase + mt * 16 + laneR;     // global X row for A fragments

  v8f acc[4];

  // ---- net = X @ WpT + bp  (K = 160) ----
#pragma unroll
  for (int t = 0; t < 4; ++t) acc[t] = vzero8();
  for (int kb = 0; kb < KPAD; kb += 32) {
    const v16h a = frag_a(Xh + (size_t)aRow * KPAD + kb + laneH * 8);
#pragma unroll
    for (int t = 0; t < 4; ++t) {
      const v16h bf = frag_b(WpT + ((ntB + t) * 16 + laneR) * KPAD + kb + laneH * 8);
      acc[t] = __builtin_amdgcn_wmma_f32_16x16x32_f16(false, a, false, bf,
                                                      (short)0, acc[t], false, false);
    }
  }
#pragma unroll
  for (int t = 0; t < 4; ++t) {
    const int c = (ntB + t) * 16 + laneR;
    const float bias = bp[c];
#pragma unroll
    for (int i = 0; i < 8; ++i) sNet[mt * 16 + i + 8 * laneH][c] = acc[t][i] + bias;
  }
  __syncthreads();

  // ---- 4 residual blocks ----
  for (int blk = 0; blk < NBLK; ++blk) {
    const float* g0b  = g0  + blk * HID;
    const float* b0b  = b0  + blk * HID;
    const float* bf0b = bf0 + blk * HID;
    const float* g1b  = g1  + blk * HID;
    const float* b1b  = b1  + blk * HID;
    const float* bf1b = bf1 + blk * HID;
    const float* bcb  = bc  + blk * HID;
    const _Float16* W0b = W0T + (size_t)blk * HID * HID;
    const _Float16* W1b = W1T + (size_t)blk * HID * HID;
    const _Float16* Wcb = WcT + (size_t)blk * HID * KPAD;

    // act = relu(bn0(net)) -> f16
    for (int idx = tid; idx < 64 * HID; idx += 256) {
      const int r = idx >> 7, c = idx & 127;
      float v = sNet[r][c] * g0b[c] + b0b[c];
      sAct[r][c] = (_Float16)(v > 0.0f ? v : 0.0f);
    }
    __syncthreads();

    // h = act @ W0 (K = 128)
#pragma unroll
    for (int t = 0; t < 4; ++t) acc[t] = vzero8();
    for (int kb = 0; kb < HID; kb += 32) {
      const v16h a = frag_a(&sAct[mt * 16 + laneR][kb + laneH * 8]);
#pragma unroll
      for (int t = 0; t < 4; ++t) {
        const v16h bf = frag_b(W0b + ((ntB + t) * 16 + laneR) * HID + kb + laneH * 8);
        acc[t] = __builtin_amdgcn_wmma_f32_16x16x32_f16(false, a, false, bf,
                                                        (short)0, acc[t], false, false);
      }
    }
    __syncthreads();   // all waves done reading sAct before overwrite

    // sAct = relu(bn1(h + bf0)) -> f16 (wave-owned tiles)
#pragma unroll
    for (int t = 0; t < 4; ++t) {
      const int c = (ntB + t) * 16 + laneR;
      const float F = bf0b[c], G = g1b[c], Bv = b1b[c];
#pragma unroll
      for (int i = 0; i < 8; ++i) {
        float v = (acc[t][i] + F) * G + Bv;
        sAct[mt * 16 + i + 8 * laneH][c] = (_Float16)(v > 0.0f ? v : 0.0f);
      }
    }
    __syncthreads();

    // dx = act2 @ W1 (K = 128)   and   ctx = X @ Wc (K = 160)
    v8f acc2[4];
#pragma unroll
    for (int t = 0; t < 4; ++t) { acc[t] = vzero8(); acc2[t] = vzero8(); }
    for (int kb = 0; kb < HID; kb += 32) {
      const v16h a = frag_a(&sAct[mt * 16 + laneR][kb + laneH * 8]);
#pragma unroll
      for (int t = 0; t < 4; ++t) {
        const v16h bf = frag_b(W1b + ((ntB + t) * 16 + laneR) * HID + kb + laneH * 8);
        acc[t] = __builtin_amdgcn_wmma_f32_16x16x32_f16(false, a, false, bf,
                                                        (short)0, acc[t], false, false);
      }
    }
    for (int kb = 0; kb < KPAD; kb += 32) {
      const v16h a = frag_a(Xh + (size_t)aRow * KPAD + kb + laneH * 8);
#pragma unroll
      for (int t = 0; t < 4; ++t) {
        const v16h bf = frag_b(Wcb + ((ntB + t) * 16 + laneR) * KPAD + kb + laneH * 8);
        acc2[t] = __builtin_amdgcn_wmma_f32_16x16x32_f16(false, a, false, bf,
                                                         (short)0, acc2[t], false, false);
      }
    }

    // net += dx + bf1 + ctx + bc (wave-owned tiles)
#pragma unroll
    for (int t = 0; t < 4; ++t) {
      const int c = (ntB + t) * 16 + laneR;
      const float add = bf1b[c] + bcb[c];
#pragma unroll
      for (int i = 0; i < 8; ++i)
        sNet[mt * 16 + i + 8 * laneH][c] += acc[t][i] + acc2[t][i] + add;
    }
    __syncthreads();
  }

  // ---- output head: relu(bnO(net)) @ Wo + bo ----
  if (tid < 64) {
    float s = bo[0];
    for (int c = 0; c < HID; ++c) {
      float v = sNet[tid][c] * gO[c] + bO[c];
      v = v > 0.0f ? v : 0.0f;
      s += v * Wo[c];
    }
    pred[rowBase + tid] = s;
  }
}

// ---------------- kernel 8: loss reduction ----------------
__global__ __launch_bounds__(256) void loss_kernel(const float* __restrict__ target,
                                                   const float* __restrict__ pred,
                                                   float* __restrict__ out) {
  __shared__ float red[256];
  float s = 0.0f;
  for (int i = threadIdx.x; i < MROWS; i += 256) {
    const float d = target[i] - pred[i];
    s += d * d;
  }
  red[threadIdx.x] = s;
  __syncthreads();
  for (int off = 128; off > 0; off >>= 1) {
    if (threadIdx.x < off) red[threadIdx.x] += red[threadIdx.x + off];
    __syncthreads();
  }
  if (threadIdx.x == 0) out[0] = 0.5f * (red[0] / (float)MROWS) * 100.0f;  // 1/DSM_SIGMA = 100
}

// ---------------- launch ----------------
extern "C" void kernel_launch(void* const* d_in, const int* in_sizes, int n_in,
                              void* d_out, int out_size, void* d_ws, size_t ws_size,
                              hipStream_t stream) {
  const float* pcl_noisy = (const float*)d_in[0];
  const float* pcl_clean = (const float*)d_in[1];
  const int*   pnt_idx   = (const int*)  d_in[2];
  const float* feW1 = (const float*)d_in[3];
  const float* feb1 = (const float*)d_in[4];
  const float* feW2 = (const float*)d_in[5];
  const float* feb2 = (const float*)d_in[6];
  const float* Wp   = (const float*)d_in[7];
  const float* bp   = (const float*)d_in[8];
  const float* g0   = (const float*)d_in[9];
  const float* b0   = (const float*)d_in[10];
  const float* W0   = (const float*)d_in[11];
  const float* bf0  = (const float*)d_in[12];
  const float* g1   = (const float*)d_in[13];
  const float* b1   = (const float*)d_in[14];
  const float* W1   = (const float*)d_in[15];
  const float* bf1  = (const float*)d_in[16];
  const float* Wc   = (const float*)d_in[17];
  const float* bc   = (const float*)d_in[18];
  const float* gO   = (const float*)d_in[19];
  const float* bO   = (const float*)d_in[20];
  const float* Wo   = (const float*)d_in[21];
  const float* bo   = (const float*)d_in[22];

  char* ws = (char*)d_ws;
  int*      knn16  = (int*)     (ws + 0);          // 512*16*4      = 32768
  float*    feat   = (float*)   (ws + 32768);      // 512*128*4     = 262144
  float*    frames = (float*)   (ws + 294912);     // 512*32*3*4    = 196608
  float*    target = (float*)   (ws + 491520);     // 16384*4       = 65536
  float*    pred   = (float*)   (ws + 557056);     // 16384*4       = 65536
  _Float16* Xh     = (_Float16*)(ws + 622592);     // 16384*160*2   = 5242880
  _Float16* WpT    = (_Float16*)(ws + 5865472);    // 128*160*2     = 40960
  _Float16* W0T    = (_Float16*)(ws + 5906432);    // 4*128*128*2   = 131072
  _Float16* W1T    = (_Float16*)(ws + 6037504);    // 4*128*128*2   = 131072
  _Float16* WcT    = (_Float16*)(ws + 6168576);    // 4*128*160*2   = 163840

  knn16_kernel   <<<BB,   128, 0, stream>>>(pcl_noisy, pnt_idx, knn16);
  featnet_kernel <<<NPTS, 128, 0, stream>>>(pcl_noisy, pnt_idx, knn16,
                                            feW1, feb1, feW2, feb2, feat);
  frameknn_kernel<<<BB,   128, 0, stream>>>(pcl_noisy, pnt_idx, frames);
  cleanknn_kernel<<<MROWS / 256, 256, 0, stream>>>(pcl_clean, frames, target);

  wt_convert<<<(HID * KPAD + 255) / 256, 256, 0, stream>>>(Wp, WpT, CDIM, HID, KPAD);
  for (int i = 0; i < NBLK; ++i) {
    wt_convert<<<(HID * HID + 255) / 256, 256, 0, stream>>>(
        W0 + (size_t)i * HID * HID, W0T + (size_t)i * HID * HID, HID, HID, HID);
    wt_convert<<<(HID * HID + 255) / 256, 256, 0, stream>>>(
        W1 + (size_t)i * HID * HID, W1T + (size_t)i * HID * HID, HID, HID, HID);
    wt_convert<<<(HID * KPAD + 255) / 256, 256, 0, stream>>>(
        Wc + (size_t)i * CDIM * HID, WcT + (size_t)i * HID * KPAD, CDIM, HID, KPAD);
  }

  pack_x_kernel<<<MROWS, 192, 0, stream>>>(pcl_noisy, pnt_idx, frames, feat, Xh);

  score_wmma_kernel<<<MROWS / 64, 256, 0, stream>>>(
      Xh, WpT, W0T, W1T, WcT,
      bp, g0, b0, bf0, g1, b1, bf1, bc, gO, bO, Wo, bo, pred);

  loss_kernel<<<1, 256, 0, stream>>>(target, pred, (float*)d_out);
}
